// GAT_AC_4475355922700
// MI455X (gfx1250) — matrix operand order repaired and checked by reading, not compiled, not run on hardware
//
#include <hip/hip_runtime.h>
#include <hip/hip_bf16.h>

typedef __attribute__((ext_vector_type(16))) _Float16 v16h;
typedef __attribute__((ext_vector_type(8)))  float    v8f;

#define NEG_SLOPE 0.2f

__device__ __forceinline__ unsigned f2s(float f) {
  unsigned u = __float_as_uint(f);
  return (u & 0x80000000u) ? ~u : (u | 0x80000000u);
}
__device__ __forceinline__ float s2f(unsigned s) {
  return __uint_as_float((s & 0x80000000u) ? (s & 0x7FFFFFFFu) : ~s);
}
__device__ __forceinline__ float elu1(float v)  { return v > 0.f ? v : __expf(v) - 1.f; }
__device__ __forceinline__ float lrelu(float v) { return v >= 0.f ? v : NEG_SLOPE * v; }

// ---- scalar attention constants: cs1[4], cd1[4], ce1[k=0][4], ce1[k=1][4], ce2[2]
__global__ void gat_const(const float* W1, const float* a_src1, const float* a_dst1,
                          const float* We1, const float* a_e1,
                          const float* We2, const float* a_e2, float* cst) {
  int t = threadIdx.x;
  if (t < 4) {
    float s = 0.f; for (int c = 0; c < 64; ++c) s += W1[t*64+c] * a_src1[t*64+c];
    cst[t] = s;
  } else if (t < 8) {
    int hd = t - 4; float s = 0.f;
    for (int c = 0; c < 64; ++c) s += W1[hd*64+c] * a_dst1[hd*64+c];
    cst[t] = s;
  } else if (t < 16) {
    int k = (t-8) >> 2, hd = (t-8) & 3; float s = 0.f;
    for (int c = 0; c < 64; ++c) s += We1[k*256 + hd*64 + c] * a_e1[hd*64+c];
    cst[t] = s;
  } else if (t < 18) {
    int k = t - 16; float s = 0.f;
    for (int c = 0; c < 64; ++c) s += We2[k*64+c] * a_e2[c];
    cst[t] = s;
  }
}

// ---- deg + segment_sum(edge_attr) over dst
__global__ void gat_degsum(const int* dst, const float* ea, float* deg, float* sumea, int E) {
  int e = blockIdx.x * blockDim.x + threadIdx.x;
  if (e >= E) return;
  int d = dst[e];
  atomicAdd(&deg[d], 1.0f);
  atomicAdd(&sumea[d*2+0], ea[e*2+0]);
  atomicAdd(&sumea[d*2+1], ea[e*2+1]);
}

__global__ void gat_loopattr(const float* deg, const float* sumea, float* loop, int N) {
  int n = blockIdx.x * blockDim.x + threadIdx.x;
  if (n >= N) return;
  float dg = fmaxf(deg[n], 1.0f);
  loop[n*2+0] = sumea[n*2+0] / dg;
  loop[n*2+1] = sumea[n*2+1] / dg;
}

// ---- conv1 edge pass A: raw leaky logits + segment max (sortable uint)
__global__ void gat_e1max(const int* src, const int* dst, const float* ea, const float* loop,
                          const float* x, const float* cst, float* araw, unsigned* amax,
                          int E, int ET) {
  int e = blockIdx.x * blockDim.x + threadIdx.x;
  if (e >= ET) return;
  int s, d; float e0, e1;
  if (e < E) { s = src[e]; d = dst[e]; e0 = ea[e*2]; e1 = ea[e*2+1]; }
  else       { s = d = e - E; e0 = loop[s*2]; e1 = loop[s*2+1]; }
  float xs = x[s], xd = x[d];
#pragma unroll
  for (int hd = 0; hd < 4; ++hd) {
    float a = xs*cst[hd] + xd*cst[4+hd] + e0*cst[8+hd] + e1*cst[12+hd];
    a = lrelu(a);
    araw[e*4+hd] = a;
    atomicMax(&amax[d*4+hd], f2s(a));
  }
}

// ---- conv1 edge pass B: exp(a - max) + segment sum
__global__ void gat_e1exp(const int* dst, float* araw, const unsigned* amax, float* den,
                          int E, int ET) {
  int e = blockIdx.x * blockDim.x + threadIdx.x;
  if (e >= ET) return;
  int d = (e < E) ? dst[e] : (e - E);
#pragma unroll
  for (int hd = 0; hd < 4; ++hd) {
    float w = __expf(araw[e*4+hd] - s2f(amax[d*4+hd]));
    araw[e*4+hd] = w;
    atomicAdd(&den[d*4+hd], w);
  }
}

// ---- conv1 edge pass C: normalized scalar aggregation S[d,h] += alpha * x[s]
__global__ void gat_e1agg(const int* src, const int* dst, const float* x, const float* araw,
                          const float* den, float* S, int E, int ET) {
  int e = blockIdx.x * blockDim.x + threadIdx.x;
  if (e >= ET) return;
  int s, d;
  if (e < E) { s = src[e]; d = dst[e]; } else { s = d = e - E; }
  float xs = x[s];
#pragma unroll
  for (int hd = 0; hd < 4; ++hd)
    atomicAdd(&S[d*4+hd], (araw[e*4+hd] / den[d*4+hd]) * xs);
}

// ---- h1[n,c] = elu(W1[c]*S[n, c/64] + b1[c]) -> f16 for WMMA
__global__ void gat_h1(const float* W1, const float* b1, const float* S, _Float16* h1, int N) {
  int idx = blockIdx.x * blockDim.x + threadIdx.x;
  if (idx >= N * 256) return;
  int n = idx >> 8, c = idx & 255, hd = c >> 6;
  h1[idx] = (_Float16)elu1(W1[c] * S[n*4+hd] + b1[c]);
}

__global__ void gat_f2h(const float* in, _Float16* out, int n) {
  int i = blockIdx.x * blockDim.x + threadIdx.x;
  if (i < n) out[i] = (_Float16)in[i];
}

// ---- transpose + f16: wb16t[col*128 + k] = Wbuild[k*2000 + col]   (K=128, Ncols=2000)
__global__ void gat_f2h_t(const float* in, _Float16* out, int K, int Ncols) {
  int idx = blockIdx.x * blockDim.x + threadIdx.x;
  if (idx >= K * Ncols) return;
  int col = idx >> 7, k = idx & 127;          // K == 128
  out[idx] = (_Float16)in[k * Ncols + col];
}

// ---- GEMM1: h2pre[40000x64] = h1[40000x256] @ W2[256x64]
// B staged in LDS pre-swizzled to per-lane-contiguous fragment layout:
//   Bs[((kk*4 + tn)*32 + lane)*16 + i] = B[(kk*32 + hi*16 + i)*64 + tn*16 + r]
// so each lane's 16-half fragment is one contiguous 32B run -> 2x ds_load_b128.
__global__ void gat_gemm1(const _Float16* __restrict__ A, const _Float16* __restrict__ B,
                          float* __restrict__ C, int Mtiles) {
  __shared__ _Float16 Bs[8 * 4 * 32 * 16];    // 16384 halfs = 32KB
  for (int idx = threadIdx.x; idx < 16384; idx += blockDim.x) {
    int i    = idx & 15;
    int ln   = (idx >> 4) & 31;
    int tnkk = idx >> 9;
    int tn   = tnkk & 3;
    int kk   = tnkk >> 2;
    int rr   = ln & 15, hh = ln >> 4;
    Bs[idx] = B[(kk*32 + hh*16 + i) * 64 + tn*16 + rr];
  }
  __syncthreads();
  int wave = (blockIdx.x * blockDim.x + threadIdx.x) >> 5;
  int lane = threadIdx.x & 31;
  int tm = wave >> 2, tn = wave & 3;
  if (tm >= Mtiles) return;                   // wave-uniform
  int r = lane & 15, hi = lane >> 4;
  int row = tm * 16 + r;
  v8f acc = {};
#pragma unroll
  for (int kk = 0; kk < 8; ++kk) {
    int k0 = kk * 32;
    v16h a, b;
    const _Float16* ap = A + (size_t)row * 256 + k0 + hi * 16;
#pragma unroll
    for (int i = 0; i < 8; ++i) a[i] = ap[i];        // K = k0 + hi*16 + 0..7
#pragma unroll
    for (int i = 0; i < 8; ++i) a[8+i] = ap[8 + i];  // K = k0 + hi*16 + 8..15
    const _Float16* bp = Bs + ((kk*4 + tn)*32 + lane) * 16;
#pragma unroll
    for (int i = 0; i < 16; ++i) b[i] = bp[i];       // contiguous -> ds_load_b128 x2
    acc = __builtin_amdgcn_wmma_f32_16x16x32_f16(false, a, false, b, (short)0, acc, false, false);
  }
  int mBase = tm * 16 + hi * 8;
  int nOut  = tn * 16 + r;
#pragma unroll
  for (int i = 0; i < 8; ++i) C[(size_t)(mBase + i) * 64 + nOut] = acc[i];
}

// ---- conv2 per-node logit halves
__global__ void gat_asad2(const float* h2, const float* a_src2, const float* a_dst2,
                          float* as2, float* ad2, int N) {
  int n = blockIdx.x * blockDim.x + threadIdx.x;
  if (n >= N) return;
  float ss = 0.f, sd = 0.f;
  for (int c = 0; c < 64; ++c) {
    float v = h2[(size_t)n*64 + c];
    ss += v * a_src2[c];
    sd += v * a_dst2[c];
  }
  as2[n] = ss; ad2[n] = sd;
}

__global__ void gat_e2max(const int* src, const int* dst, const float* ea, const float* loop,
                          const float* as2, const float* ad2, const float* cst,
                          float* wbuf, unsigned* amax, int E, int ET) {
  int e = blockIdx.x * blockDim.x + threadIdx.x;
  if (e >= ET) return;
  int s, d; float e0, e1;
  if (e < E) { s = src[e]; d = dst[e]; e0 = ea[e*2]; e1 = ea[e*2+1]; }
  else       { s = d = e - E; e0 = loop[s*2]; e1 = loop[s*2+1]; }
  float a = lrelu(as2[s] + ad2[d] + e0*cst[16] + e1*cst[17]);
  wbuf[e] = a;
  atomicMax(&amax[d], f2s(a));
}

__global__ void gat_e2exp(const int* dst, float* wbuf, const unsigned* amax, float* den,
                          int E, int ET) {
  int e = blockIdx.x * blockDim.x + threadIdx.x;
  if (e >= ET) return;
  int d = (e < E) ? dst[e] : (e - E);
  float w = __expf(wbuf[e] - s2f(amax[d]));
  wbuf[e] = w;
  atomicAdd(&den[d], w);
}

// ---- conv2 scatter aggregate: out2[d,:] += alpha * h2pre[s,:]   (ET*16 threads, 4 ch each)
__global__ void gat_e2agg(const int* src, const int* dst, const float* wbuf, const float* den,
                          const float* h2, float* out2, int E, int ET) {
  int t = blockIdx.x * blockDim.x + threadIdx.x;
  if (t >= ET * 16) return;
  int e = t >> 4, c0 = (t & 15) << 2;
  int s, d;
  if (e < E) { s = src[e]; d = dst[e]; } else { s = d = e - E; }
  float coeff = wbuf[e] / den[d];
  const float4 hv = *(const float4*)(h2 + (size_t)s*64 + c0);
  atomicAdd(&out2[(size_t)d*64 + c0 + 0], coeff * hv.x);
  atomicAdd(&out2[(size_t)d*64 + c0 + 1], coeff * hv.y);
  atomicAdd(&out2[(size_t)d*64 + c0 + 2], coeff * hv.z);
  atomicAdd(&out2[(size_t)d*64 + c0 + 3], coeff * hv.w);
}

// ---- h2 = elu(out2 + b2); mean-pool into g[G,64]
__global__ void gat_h2pool(const float* out2, const float* b2, const int* batch,
                           float* g, int N, float invCnt) {
  int idx = blockIdx.x * blockDim.x + threadIdx.x;
  if (idx >= N * 64) return;
  int n = idx >> 6, c = idx & 63;
  float v = elu1(out2[idx] + b2[c]);
  atomicAdd(&g[batch[n]*64 + c], v * invCnt);
}

// ---- lambda MLP
__global__ void gat_lam(const float* lv, const float* lpW1, const float* lpb1,
                        const float* lpW2, const float* lpb2, float* lam, int G) {
  int t = blockIdx.x * blockDim.x + threadIdx.x;
  if (t >= G * 64) return;
  int i = t >> 6, c2 = t & 63;
  float L = lv[i], s = lpb2[c2];
  for (int c = 0; c < 64; ++c) {
    float t1 = fmaxf(L * lpW1[c] + lpb1[c], 0.f);
    s += t1 * lpW2[c*64 + c2];
  }
  lam[t] = s;
}

// ---- gf16: 32x128 f16, rows 0..19 = [g | lam], rows 20..31 zero
__global__ void gat_gf16(const float* g, const float* lam, _Float16* gf16) {
  int t = blockIdx.x * blockDim.x + threadIdx.x;
  if (t >= 32 * 128) return;
  int i = t >> 7, k = t & 127;
  float v = 0.f;
  if (i < 20) v = (k < 64) ? g[i*64 + k] : lam[i*64 + (k - 64)];
  gf16[t] = (_Float16)v;
}

// ---- GEMM2: lb[20x2000] = gf[32x128] @ Wbuild[128x2000] + bbuild
// B is pre-transposed (wb16t[col*128 + k]) so lane fragments are contiguous.
__global__ void gat_gemm2(const _Float16* __restrict__ A, const _Float16* __restrict__ Bt,
                          const float* __restrict__ bias, float* __restrict__ out) {
  int wave = (blockIdx.x * blockDim.x + threadIdx.x) >> 5;   // 0..249
  int lane = threadIdx.x & 31;
  int r = lane & 15, hi = lane >> 4;
  int tm = wave / 125, tn = wave % 125;
  int row = tm * 16 + r;
  int col = tn * 16 + r;
  v8f acc = {};
#pragma unroll
  for (int kk = 0; kk < 4; ++kk) {
    int k0 = kk * 32;
    v16h a, b;
    const _Float16* ap = A + row * 128 + k0 + hi * 16;
#pragma unroll
    for (int i = 0; i < 8; ++i) a[i] = ap[i];
#pragma unroll
    for (int i = 0; i < 8; ++i) a[8+i] = ap[8 + i];
    const _Float16* bp = Bt + col * 128 + k0 + hi * 16;      // contiguous 16 halfs
#pragma unroll
    for (int i = 0; i < 16; ++i) b[i] = bp[i];               // -> global_load_b128 x2
    acc = __builtin_amdgcn_wmma_f32_16x16x32_f16(false, a, false, b, (short)0, acc, false, false);
  }
  int mBase = tm * 16 + hi * 8;
  int nOut  = tn * 16 + r;
  float bv = bias[nOut];
#pragma unroll
  for (int i = 0; i < 8; ++i) {
    int m = mBase + i;
    if (m < 20) out[m * 2000 + nOut] = acc[i] + bv;
  }
}

// ---- small heads: trunk(5) branch(5) micro(3) value(1)
__global__ void gat_heads(const float* g, const float* lam,
                          const float* Wt, const float* bt,
                          const float* Wb, const float* bb,
                          const float* Wm, const float* bm,
                          const float* Wv, const float* bv,
                          float* out) {
  int t = blockIdx.x * blockDim.x + threadIdx.x;
  if (t >= 280) return;
  int i, j, nc, base; const float* W; const float* b;
  if (t < 100)      { i = t / 5;        j = t % 5;        nc = 5; W = Wt; b = bt; base = 40000 + t; }
  else if (t < 200) { int u = t - 100; i = u / 5; j = u % 5; nc = 5; W = Wb; b = bb; base = 40100 + u; }
  else if (t < 260) { int u = t - 200; i = u / 3; j = u % 3; nc = 3; W = Wm; b = bm; base = 40200 + u; }
  else              { i = t - 260;      j = 0;             nc = 1; W = Wv; b = bv; base = 40260 + (t - 260); }
  float s = b[j];
  for (int k = 0; k < 128; ++k) {
    float gf = (k < 64) ? g[i*64 + k] : lam[i*64 + (k - 64)];
    s += gf * W[k*nc + j];
  }
  out[base] = s;
}

extern "C" void kernel_launch(void* const* d_in, const int* in_sizes, int n_in,
                              void* d_out, int out_size, void* d_ws, size_t ws_size,
                              hipStream_t stream) {
  const int N  = in_sizes[0];          // 40000
  const int E  = in_sizes[1] / 2;      // 640000
  const int ET = E + N;
  const int G  = in_sizes[4];          // 20

  const float* x    = (const float*)d_in[0];
  const int*   src  = (const int*)d_in[1];
  const int*   dst  = src + E;
  const float* ea   = (const float*)d_in[2];
  const int*   batch= (const int*)d_in[3];
  const float* lv   = (const float*)d_in[4];
  const float* W1   = (const float*)d_in[5];
  const float* a_s1 = (const float*)d_in[6];
  const float* a_d1 = (const float*)d_in[7];
  const float* We1  = (const float*)d_in[8];
  const float* a_e1 = (const float*)d_in[9];
  const float* b1   = (const float*)d_in[10];
  const float* W2   = (const float*)d_in[11];
  const float* a_s2 = (const float*)d_in[12];
  const float* a_d2 = (const float*)d_in[13];
  const float* We2  = (const float*)d_in[14];
  const float* a_e2 = (const float*)d_in[15];
  const float* b2   = (const float*)d_in[16];
  const float* lpW1 = (const float*)d_in[17];
  const float* lpb1 = (const float*)d_in[18];
  const float* lpW2 = (const float*)d_in[19];
  const float* lpb2 = (const float*)d_in[20];
  const float* Wbuild = (const float*)d_in[21];
  const float* bbuild = (const float*)d_in[22];
  const float* Wtr  = (const float*)d_in[23];
  const float* btr  = (const float*)d_in[24];
  const float* Wbr  = (const float*)d_in[25];
  const float* bbr  = (const float*)d_in[26];
  const float* Wmi  = (const float*)d_in[27];
  const float* bmi  = (const float*)d_in[28];
  const float* Wvl  = (const float*)d_in[29];
  const float* bvl  = (const float*)d_in[30];
  float* out = (float*)d_out;

  // ---- workspace layout (all sub-blocks 256B-aligned)
  char* w = (char*)d_ws;
  size_t off = 0;
  auto alloc = [&](size_t bytes) -> void* {
    void* p = w + off;
    off = (off + bytes + 255) & ~(size_t)255;
    return p;
  };
  // zero-init (one contiguous memset) ---
  float*    deg   = (float*)alloc((size_t)N * 4);
  float*    sumea = (float*)alloc((size_t)N * 2 * 4);
  unsigned* amax1 = (unsigned*)alloc((size_t)N * 4 * 4);
  float*    den1  = (float*)alloc((size_t)N * 4 * 4);
  float*    S     = (float*)alloc((size_t)N * 4 * 4);
  unsigned* amax2 = (unsigned*)alloc((size_t)N * 4);
  float*    den2  = (float*)alloc((size_t)N * 4);
  float*    g     = (float*)alloc((size_t)G * 64 * 4);
  float*    out2  = (float*)alloc((size_t)N * 64 * 4);
  size_t zeroBytes = off;
  // non-zeroed ---
  float*    loop  = (float*)alloc((size_t)N * 2 * 4);
  float*    araw  = (float*)alloc((size_t)ET * 4 * 4);
  _Float16* h1    = (_Float16*)alloc((size_t)N * 256 * 2);
  _Float16* w2h   = (_Float16*)alloc((size_t)256 * 64 * 2);
  float*    h2pre = (float*)alloc((size_t)N * 64 * 4);
  float*    as2   = (float*)alloc((size_t)N * 4);
  float*    ad2   = (float*)alloc((size_t)N * 4);
  float*    wbuf  = (float*)alloc((size_t)ET * 4);
  float*    lam   = (float*)alloc((size_t)G * 64 * 4);
  _Float16* gf16  = (_Float16*)alloc((size_t)32 * 128 * 2);
  _Float16* wb16t = (_Float16*)alloc((size_t)128 * 2000 * 2);
  float*    cst   = (float*)alloc(32 * 4);
  (void)ws_size;

  auto nb = [](long n, int b) { return (int)((n + b - 1) / b); };

  hipMemsetAsync(d_ws, 0, zeroBytes, stream);
  gat_const<<<1, 32, 0, stream>>>(W1, a_s1, a_d1, We1, a_e1, We2, a_e2, cst);
  gat_degsum<<<nb(E, 256), 256, 0, stream>>>(dst, ea, deg, sumea, E);
  gat_loopattr<<<nb(N, 256), 256, 0, stream>>>(deg, sumea, loop, N);

  // conv1 (scalar-attention form)
  gat_e1max<<<nb(ET, 256), 256, 0, stream>>>(src, dst, ea, loop, x, cst, araw, amax1, E, ET);
  gat_e1exp<<<nb(ET, 256), 256, 0, stream>>>(dst, araw, amax1, den1, E, ET);
  gat_e1agg<<<nb(ET, 256), 256, 0, stream>>>(src, dst, x, araw, den1, S, E, ET);
  gat_h1<<<nb((long)N * 256, 256), 256, 0, stream>>>(W1, b1, S, h1, N);

  // GEMM1: h1 @ W2 via WMMA (B swizzled into LDS fragments)
  gat_f2h<<<nb(256 * 64, 256), 256, 0, stream>>>(W2, w2h, 256 * 64);
  gat_gemm1<<<(N / 16) * 4 / 8, 256, 0, stream>>>(h1, w2h, h2pre, N / 16);

  // conv2 attention + scatter aggregate
  gat_asad2<<<nb(N, 256), 256, 0, stream>>>(h2pre, a_s2, a_d2, as2, ad2, N);
  gat_e2max<<<nb(ET, 256), 256, 0, stream>>>(src, dst, ea, loop, as2, ad2, cst, wbuf, amax2, E, ET);
  gat_e2exp<<<nb(ET, 256), 256, 0, stream>>>(dst, wbuf, amax2, den2, E, ET);
  gat_e2agg<<<nb((long)ET * 16, 256), 256, 0, stream>>>(src, dst, wbuf, den2, h2pre, out2, E, ET);
  gat_h2pool<<<nb((long)N * 64, 256), 256, 0, stream>>>(out2, b2, batch, g, N, (float)G / (float)N);

  // lambda MLP + gf assembly
  gat_lam<<<nb(G * 64, 64), 64, 0, stream>>>(lv, lpW1, lpb1, lpW2, lpb2, lam, G);
  gat_gf16<<<16, 256, 0, stream>>>(g, lam, gf16);

  // GEMM2: gf @ Wbuild (+bias) via WMMA, Wbuild pre-transposed to col-major f16
  gat_f2h_t<<<nb(128 * 2000, 256), 256, 0, stream>>>(Wbuild, wb16t, 128, 2000);
  gat_gemm2<<<125, 64, 0, stream>>>(gf16, wb16t, bbuild, out);

  // small heads -> d_out[40000..40279]
  gat_heads<<<1, 288, 0, stream>>>(g, lam, Wtr, btr, Wbr, bbr, Wmi, bmi, Wvl, bvl, out);
}
